// Model_84550726189303
// MI455X (gfx1250) — matrix operand order repaired and checked by reading, not compiled, not run on hardware
//
#include <hip/hip_runtime.h>
#include <hip/hip_bf16.h>

// ---------- WMMA vector types (CDNA5 / gfx1250, wave32) ----------
typedef __attribute__((ext_vector_type(16))) __bf16       v16bf;
typedef __attribute__((ext_vector_type(8)))  float        v8f;
typedef __attribute__((ext_vector_type(8)))  unsigned int v8u;

union F8 { v8f v; float s[8]; };
union U8 { v8u v; unsigned int s[8]; };

// A-fragment (16x32 bf16): lane holds row (lane&15), K in
// [khalf, khalf+8) U [khalf+16, khalf+24), khalf = (lane>>4)*8.
__device__ __forceinline__ v16bf load_a_frag(const __bf16* row, int khalf) {
  U8 u;
#pragma unroll
  for (int i = 0; i < 4; ++i)
    u.s[i] = *(const unsigned int*)(row + khalf + 2 * i);
#pragma unroll
  for (int i = 0; i < 4; ++i)
    u.s[4 + i] = *(const unsigned int*)(row + khalf + 16 + 2 * i);
  return __builtin_bit_cast(v16bf, u.v);
}

// B-fragment (32x16 bf16): lane holds col (lane&15), K in
// [kh, kh+8) U [kh+16, kh+24), kh = khalfB/2 = (lane>>4)*8; W stored K-contiguous (transposed).
__device__ __forceinline__ v16bf load_b_frag(const __bf16* col, int khalfB) {
  U8 u;
#pragma unroll
  for (int i = 0; i < 8; ++i)
    u.s[i] = *(const unsigned int*)(col + (khalfB >> 1) + (i >> 2) * 16 + 2 * (i & 3));
  return __builtin_bit_cast(v16bf, u.v);
}

__device__ __forceinline__ unsigned lds_off(const void* p) {
  // LDS aperture: low 32 bits of the flat address are the LDS byte address.
  return (unsigned)(uintptr_t)p;
}

// ================= Kernel 1: fused input projection =================
// out[M,64] += X[M,Kslice] @ W[Kslice,64]  (+ b + emb on slice 0).
// Split-K across gridDim.y for occupancy; f32 atomics into pre-zeroed out.
// A tile register-staged; W tile staged raw via the gfx1250 async engine.
__global__ __launch_bounds__(256) void gemm_in_kernel(
    const float* __restrict__ X, const float* __restrict__ W,
    const float* __restrict__ b, const float* __restrict__ emb,
    float* __restrict__ out, int K) {
  __shared__ __bf16 sA[128][72];                 // 128 x 64 bf16 (+pad)
  __shared__ __bf16 sBt[64][72];                 // W^T chunk, bf16
  __shared__ alignas(16) float sWraw[64 * 64];   // async-staged fp32 W chunk

  const int tid = threadIdx.x;
  const int warp = tid >> 5, lane = tid & 31;
  const int rowBase = blockIdx.x * 128;
  const int m = lane & 15;
  const int khalfA = (lane >> 4) * 8;
  const int khalfB = (lane >> 4) * 16;
  const int kLen = K / gridDim.y;                // K slice per y-block
  const int kStart = blockIdx.y * kLen;

  F8 acc[4];
#pragma unroll
  for (int t = 0; t < 4; ++t) acc[t].v = (v8f){0.f,0.f,0.f,0.f,0.f,0.f,0.f,0.f};

  for (int k0 = kStart; k0 < kStart + kLen; k0 += 64) {
    // ---- issue all A-tile loads (128x64 fp32 = 2048 float4, 8/thread) ----
    float4 fa[8];
#pragma unroll
    for (int i = 0; i < 8; ++i) {
      int idx = tid + i * 256;
      int r = idx >> 4, c4 = (idx & 15) * 4;
      fa[i] = *reinterpret_cast<const float4*>(X + (size_t)(rowBase + r) * K + k0 + c4);
    }
    // ---- async-stage W chunk (64x64 fp32) into LDS (raw) ----
#pragma unroll
    for (int i = 0; i < 4; ++i) {
      int idx = tid + i * 256;                 // float4 slot
      int kk = idx >> 4, c4 = (idx & 15) * 4;
      unsigned l = lds_off(&sWraw[idx * 4]);
      const float* g = W + (size_t)(k0 + kk) * 64 + c4;
      asm volatile("global_load_async_to_lds_b128 %0, %1, off"
                   :: "v"(l), "v"((unsigned long long)(uintptr_t)g)
                   : "memory");
    }
    asm volatile("s_wait_asynccnt 0x0" ::: "memory");
    __syncthreads();   // prev chunk's WMMAs done; every wave's asyncs landed

    // ---- convert A regs -> bf16 LDS ----
#pragma unroll
    for (int i = 0; i < 8; ++i) {
      int idx = tid + i * 256;
      int r = idx >> 4, c4 = (idx & 15) * 4;
      sA[r][c4 + 0] = (__bf16)fa[i].x; sA[r][c4 + 1] = (__bf16)fa[i].y;
      sA[r][c4 + 2] = (__bf16)fa[i].z; sA[r][c4 + 3] = (__bf16)fa[i].w;
    }
    // ---- convert raw W -> bf16 transposed ----
#pragma unroll
    for (int i = 0; i < 16; ++i) {
      int idx = tid + i * 256;
      int kk = idx >> 6, n = idx & 63;
      sBt[n][kk] = (__bf16)sWraw[kk * 64 + n];
    }
    if (k0 + 64 < kStart + kLen)   // hint next A chunk -> global_prefetch_b8
      __builtin_prefetch(X + (size_t)(rowBase + warp * 16 + m) * K + k0 + 64, 0, 1);
    __syncthreads();

    // ---- preload all fragments, then 8 back-to-back WMMAs ----
    const __bf16* Arow = &sA[warp * 16 + m][0];
    v16bf af0 = load_a_frag(Arow, khalfA);
    v16bf af1 = load_a_frag(Arow + 32, khalfA);
    v16bf bfr[8];
#pragma unroll
    for (int nt = 0; nt < 4; ++nt) {
      bfr[nt]     = load_b_frag(&sBt[nt * 16 + m][0], khalfB);
      bfr[4 + nt] = load_b_frag(&sBt[nt * 16 + m][32], khalfB);
    }
#pragma unroll
    for (int nt = 0; nt < 4; ++nt)
      acc[nt].v = __builtin_amdgcn_wmma_f32_16x16x32_bf16(
          false, af0, false, bfr[nt], (short)0, acc[nt].v, false, false);
#pragma unroll
    for (int nt = 0; nt < 4; ++nt)
      acc[nt].v = __builtin_amdgcn_wmma_f32_16x16x32_bf16(
          false, af1, false, bfr[4 + nt], (short)0, acc[nt].v, false, false);
  }

  // Epilogue: vgpr r holds M=(lane>=16)*8+r, N=lane&15; atomically fold slice.
  const int g = lane >> 4, n = lane & 15;
  const bool lead = (blockIdx.y == 0);
#pragma unroll
  for (int nt = 0; nt < 4; ++nt)
#pragma unroll
    for (int r = 0; r < 8; ++r) {
      int row = rowBase + warp * 16 + g * 8 + r;
      int col = nt * 16 + n;
      float vv = acc[nt].s[r];
      if (lead) vv += b[col] + emb[(size_t)row * 64 + col];
      unsafeAtomicAdd(&out[(size_t)row * 64 + col], vv);
    }
}

// ============ Kernel 2: edge scatter-add (mean numerator) ============
// 32 edges per wave batch: coalesced index loads, __shfl broadcast.
// Feature tables + accumulators are L2-resident (192MB L2); f32 atomics.
__global__ __launch_bounds__(256) void scatter_kernel(
    const float* __restrict__ x, const int* __restrict__ src,
    const int* __restrict__ dst, float* __restrict__ msum,
    float* __restrict__ cnt, int nE) {
  const int lane = threadIdx.x & 31;
  int gw = (blockIdx.x * 256 + threadIdx.x) >> 5;
  int nW = (gridDim.x * 256) >> 5;
  for (int base = gw * 32; base < nE; base += nW * 32) {
    int e = base + lane;
    int sv = 0, dv = 0;
    if (e < nE) { sv = src[e]; dv = dst[e]; }
    int lim = nE - base; if (lim > 32) lim = 32;
    for (int j = 0; j < lim; ++j) {
      int s = __shfl(sv, j, 32);
      int d = __shfl(dv, j, 32);
      const float2 v = *reinterpret_cast<const float2*>(x + (size_t)s * 64 + lane * 2);
      unsafeAtomicAdd(&msum[(size_t)d * 64 + lane * 2 + 0], v.x);
      unsafeAtomicAdd(&msum[(size_t)d * 64 + lane * 2 + 1], v.y);
      if (lane == j) unsafeAtomicAdd(&cnt[d], 1.0f);
    }
  }
}

// ================= Kernel 3: SAGE combine =================
// out = act( (msum/max(cnt,1)) @ Wl + bl + xdst @ Wr ), H=64, bf16 WMMA.
__global__ __launch_bounds__(256) void sage_post_kernel(
    const float* __restrict__ msum, const float* __restrict__ cnt,
    const float* __restrict__ xdst, const float* __restrict__ Wl,
    const float* __restrict__ bl, const float* __restrict__ Wr,
    float* __restrict__ out, int relu) {
  __shared__ __bf16 sM[128][72];
  __shared__ __bf16 sX[128][72];
  __shared__ __bf16 sWlT[64][72];
  __shared__ __bf16 sWrT[64][72];

  const int tid = threadIdx.x;
  const int warp = tid >> 5, lane = tid & 31;
  const int rowBase = blockIdx.x * 128;

  // Stage rows: issue all global loads first, convert after.
  float4 mm[8], xx[8];
  float cv[8];
#pragma unroll
  for (int i = 0; i < 8; ++i) {
    int idx = tid + i * 256;
    int r = idx >> 4, c4 = (idx & 15) * 4;
    int row = rowBase + r;
    mm[i] = *reinterpret_cast<const float4*>(msum + (size_t)row * 64 + c4);
    xx[i] = *reinterpret_cast<const float4*>(xdst + (size_t)row * 64 + c4);
    cv[i] = cnt[row];
  }
  // Stage both weight matrices transposed (bf16)
#pragma unroll
  for (int i = 0; i < 16; ++i) {
    int idx = tid + i * 256;
    int k = idx >> 6, n = idx & 63;
    sWlT[n][k] = (__bf16)Wl[k * 64 + n];
    sWrT[n][k] = (__bf16)Wr[k * 64 + n];
  }
#pragma unroll
  for (int i = 0; i < 8; ++i) {
    int idx = tid + i * 256;
    int r = idx >> 4, c4 = (idx & 15) * 4;
    float iv = 1.0f / fmaxf(cv[i], 1.0f);
    sM[r][c4 + 0] = (__bf16)(mm[i].x * iv); sM[r][c4 + 1] = (__bf16)(mm[i].y * iv);
    sM[r][c4 + 2] = (__bf16)(mm[i].z * iv); sM[r][c4 + 3] = (__bf16)(mm[i].w * iv);
    sX[r][c4 + 0] = (__bf16)xx[i].x; sX[r][c4 + 1] = (__bf16)xx[i].y;
    sX[r][c4 + 2] = (__bf16)xx[i].z; sX[r][c4 + 3] = (__bf16)xx[i].w;
  }
  __syncthreads();

  const int m = lane & 15;
  const int khalfA = (lane >> 4) * 8;
  const int khalfB = (lane >> 4) * 16;

  F8 acc[4];
#pragma unroll
  for (int t = 0; t < 4; ++t) acc[t].v = (v8f){0.f,0.f,0.f,0.f,0.f,0.f,0.f,0.f};

#pragma unroll
  for (int pass = 0; pass < 2; ++pass) {
    const __bf16* Arow = (pass == 0) ? &sM[warp * 16 + m][0] : &sX[warp * 16 + m][0];
    // preload fragments for this operand pair, then 8 WMMAs back-to-back
    v16bf af0 = load_a_frag(Arow, khalfA);
    v16bf af1 = load_a_frag(Arow + 32, khalfA);
    v16bf bfr[8];
#pragma unroll
    for (int nt = 0; nt < 4; ++nt) {
      const __bf16* Bcol = (pass == 0) ? &sWlT[nt * 16 + m][0] : &sWrT[nt * 16 + m][0];
      bfr[nt]     = load_b_frag(Bcol, khalfB);
      bfr[4 + nt] = load_b_frag(Bcol + 32, khalfB);
    }
#pragma unroll
    for (int nt = 0; nt < 4; ++nt)
      acc[nt].v = __builtin_amdgcn_wmma_f32_16x16x32_bf16(
          false, af0, false, bfr[nt], (short)0, acc[nt].v, false, false);
#pragma unroll
    for (int nt = 0; nt < 4; ++nt)
      acc[nt].v = __builtin_amdgcn_wmma_f32_16x16x32_bf16(
          false, af1, false, bfr[4 + nt], (short)0, acc[nt].v, false, false);
  }

  const int g = lane >> 4, n = lane & 15;
#pragma unroll
  for (int nt = 0; nt < 4; ++nt)
#pragma unroll
    for (int r = 0; r < 8; ++r) {
      int row = rowBase + warp * 16 + g * 8 + r;
      int col = nt * 16 + n;
      float vv = acc[nt].s[r] + bl[col];
      if (relu) vv = fmaxf(vv, 0.0f);
      out[(size_t)row * 64 + col] = vv;
    }
}

// ============ Kernel 4: edge dot-product decoder ============
// 32 edges per wave batch; butterfly reduction; coalesced output store.
__global__ __launch_bounds__(256) void dot_kernel(
    const float* __restrict__ hc2, const float* __restrict__ hp2,
    const int* __restrict__ es, const int* __restrict__ ed,
    float* __restrict__ out, int nE) {
  const int lane = threadIdx.x & 31;
  int gw = (blockIdx.x * 256 + threadIdx.x) >> 5;
  int nW = (gridDim.x * 256) >> 5;
  for (int base = gw * 32; base < nE; base += nW * 32) {
    int e = base + lane;
    int sv = 0, dv = 0;
    if (e < nE) { sv = es[e]; dv = ed[e]; }
    int lim = nE - base; if (lim > 32) lim = 32;
    float res = 0.0f;
    for (int j = 0; j < lim; ++j) {
      int s = __shfl(sv, j, 32);
      int d = __shfl(dv, j, 32);
      const float2 a = *reinterpret_cast<const float2*>(hc2 + (size_t)s * 64 + lane * 2);
      const float2 b = *reinterpret_cast<const float2*>(hp2 + (size_t)d * 64 + lane * 2);
      float p = a.x * b.x + a.y * b.y;
#pragma unroll
      for (int off = 16; off > 0; off >>= 1) p += __shfl_xor(p, off, 32);
      if (lane == j) res = p;   // butterfly leaves total in all lanes
    }
    if (e < nE) out[e] = res;
  }
}

// ======================= launcher =======================
extern "C" void kernel_launch(void* const* d_in, const int* in_sizes, int n_in,
                              void* d_out, int out_size, void* d_ws, size_t ws_size,
                              hipStream_t stream) {
  constexpr int NC = 4096, NP = 8192, H = 64;
  const float* x_country = (const float*)d_in[0];
  const float* x_product = (const float*)d_in[1];
  const int* exp_src = (const int*)d_in[4];
  const int* exp_dst = (const int*)d_in[5];
  const int* el_src  = (const int*)d_in[6];
  const int* el_dst  = (const int*)d_in[7];
  const float* Wc = (const float*)d_in[8];
  const float* bc = (const float*)d_in[9];
  const float* Wp = (const float*)d_in[10];
  const float* bp = (const float*)d_in[11];
  const float* emb_c = (const float*)d_in[12];
  const float* emb_p = (const float*)d_in[13];
  const float* Wl_l1_exp = (const float*)d_in[14];
  const float* bl_l1_exp = (const float*)d_in[15];
  const float* Wr_l1_exp = (const float*)d_in[16];
  const float* Wl_l1_rev = (const float*)d_in[17];
  const float* bl_l1_rev = (const float*)d_in[18];
  const float* Wr_l1_rev = (const float*)d_in[19];
  const float* Wl_l2_exp = (const float*)d_in[20];
  const float* bl_l2_exp = (const float*)d_in[21];
  const float* Wr_l2_exp = (const float*)d_in[22];
  const float* Wl_l2_rev = (const float*)d_in[23];
  const float* bl_l2_rev = (const float*)d_in[24];
  const float* Wr_l2_rev = (const float*)d_in[25];
  const int E  = in_sizes[4];
  const int EL = in_sizes[6];
  float* outF = (float*)d_out;

  // workspace layout (floats)
  float* ws = (float*)d_ws;
  size_t o = 0;
  float* hc    = ws + o; o += (size_t)NC * H;
  float* hp    = ws + o; o += (size_t)NP * H;
  float* hc1   = ws + o; o += (size_t)NC * H;
  float* hp1   = ws + o; o += (size_t)NP * H;
  float* hc2   = ws + o; o += (size_t)NC * H;
  float* hp2   = ws + o; o += (size_t)NP * H;
  float* sum_p = ws + o; o += (size_t)NP * H;
  float* cnt_p = ws + o; o += (size_t)NP;
  float* sum_c = ws + o; o += (size_t)NC * H;
  float* cnt_c = ws + o; o += (size_t)NC;

  const size_t bytes_p = (size_t)(NP * H + NP) * sizeof(float); // sum_p + cnt_p
  const size_t bytes_c = (size_t)(NC * H + NC) * sizeof(float); // sum_c + cnt_c

  // ---- input projections (WMMA, split-K x4 into pre-zeroed outputs) ----
  hipMemsetAsync(hc, 0, (size_t)(NC + NP) * H * sizeof(float), stream); // hc+hp contiguous
  gemm_in_kernel<<<dim3(NC / 128, 4), 256, 0, stream>>>(x_country, Wc, bc, emb_c, hc, NC);
  gemm_in_kernel<<<dim3(NP / 128, 4), 256, 0, stream>>>(x_product, Wp, bp, emb_p, hp, NP);

  // ---- layer 1 aggregation ----
  hipMemsetAsync(sum_p, 0, bytes_p, stream);
  scatter_kernel<<<2048, 256, 0, stream>>>(hc, exp_src, exp_dst, sum_p, cnt_p, E);
  hipMemsetAsync(sum_c, 0, bytes_c, stream);
  scatter_kernel<<<2048, 256, 0, stream>>>(hp, exp_dst, exp_src, sum_c, cnt_c, E);

  // ---- layer 1 combine (+relu) ----
  sage_post_kernel<<<NP / 128, 256, 0, stream>>>(sum_p, cnt_p, hp, Wl_l1_exp,
                                                 bl_l1_exp, Wr_l1_exp, hp1, 1);
  sage_post_kernel<<<NC / 128, 256, 0, stream>>>(sum_c, cnt_c, hc, Wl_l1_rev,
                                                 bl_l1_rev, Wr_l1_rev, hc1, 1);

  // ---- layer 2 aggregation ----
  hipMemsetAsync(sum_p, 0, bytes_p, stream);
  scatter_kernel<<<2048, 256, 0, stream>>>(hc1, exp_src, exp_dst, sum_p, cnt_p, E);
  hipMemsetAsync(sum_c, 0, bytes_c, stream);
  scatter_kernel<<<2048, 256, 0, stream>>>(hp1, exp_dst, exp_src, sum_c, cnt_c, E);

  // ---- layer 2 combine (no relu) ----
  sage_post_kernel<<<NP / 128, 256, 0, stream>>>(sum_p, cnt_p, hp1, Wl_l2_exp,
                                                 bl_l2_exp, Wr_l2_exp, hp2, 0);
  sage_post_kernel<<<NC / 128, 256, 0, stream>>>(sum_c, cnt_c, hc1, Wl_l2_rev,
                                                 bl_l2_rev, Wr_l2_rev, hc2, 0);

  // ---- decoder ----
  dot_kernel<<<2048, 256, 0, stream>>>(hc2, hp2, el_src, el_dst, outF, EL);
  (void)n_in; (void)out_size; (void)ws_size;
}
